// iSTFT_82188494176784
// MI455X (gfx1250) — matrix operand order repaired and checked by reading, not compiled, not run on hardware
//
#include <hip/hip_runtime.h>

typedef __attribute__((ext_vector_type(16))) _Float16 v16h;
typedef __attribute__((ext_vector_type(8)))  float    v8f;
typedef __attribute__((ext_vector_type(4)))  float    v4f;

#define BB      16
#define FF      257
#define TT      4000
#define NFFT    512
#define HOP     256
#define OUT_LEN ((TT - 1) * HOP + NFFT)        // 1024256
#define KTOT    514                            // 257 real + 257 imag bins
#define KSTEPS  17                             // K padded to 544 = 17*32
#define EPS_F   1.1920928955078125e-07f
#define TWO_PI_OVER_N 0.01227184630308513f     // 2*pi/512

// ---------------------------------------------------------------------------
// Kernel 0: zero the output (atomics accumulate into it)
// ---------------------------------------------------------------------------
__global__ __launch_bounds__(256) void zero_kernel(v4f* __restrict__ out) {
    out[(size_t)blockIdx.x * 256 + threadIdx.x] = (v4f){0.f, 0.f, 0.f, 0.f};
}

// ---------------------------------------------------------------------------
// Kernel 1: build the f16 iDFT basis (window folded in), pre-swizzled into
// WMMA B-fragment layout:  basis[((ntile*KSTEPS + kstep)*32 + lane)*16 + h]
// where K = kstep*32 + (h&7) + 16*(h>=8) + 8*(lane>=16),  N = ntile*16 + (lane&15).
// Total = 32 * 17 * 512 = 278528 halves (~557 KB in d_ws).
// ---------------------------------------------------------------------------
__global__ __launch_bounds__(256) void basis_kernel(const float* __restrict__ window,
                                                    _Float16* __restrict__ basis) {
    int idx   = blockIdx.x * 256 + threadIdx.x;       // exact grid, no guard
    int h     = idx & 15;
    int lane  = (idx >> 4) & 31;
    int tmp   = idx >> 9;                             // 0..543
    int kstep = tmp % KSTEPS;
    int ntile = tmp / KSTEPS;
    int k     = kstep * 32 + (h & 7) + ((h & 8) ? 16 : 0) + ((lane & 16) ? 8 : 0);
    int n     = ntile * 16 + (lane & 15);

    float val = 0.f;
    if (k < FF) {                                     // real part: w_f*cos/N
        int f  = k;
        float wk = (f == 0 || f == 256) ? 1.f : 2.f;
        int ph = (f * n) & (NFFT - 1);                // exact phase reduction
        val = wk * __cosf((float)ph * TWO_PI_OVER_N) * (1.f / NFFT);
    } else if (k < KTOT) {                            // imag part: -w_f*sin/N
        int f = k - FF;
        if (f != 0 && f != 256) {
            int ph = (f * n) & (NFFT - 1);
            val = -2.f * __sinf((float)ph * TWO_PI_OVER_N) * (1.f / NFFT);
        }
    }
    val *= window[n];                                 // fold synthesis window
    basis[idx] = (_Float16)val;
}

// ---------------------------------------------------------------------------
// Kernel 2: WMMA GEMM  frames[m, n] = A[m, k] * basis[k, n]  + OLA scatter.
// One block per 16-row M tile. Since 4000 % 16 == 0, a tile never crosses a
// batch boundary: b and t_base are block-uniform. A is loaded as coalesced
// float2 (real,imag) — each 128B line of the 131.6 MB stream read exactly
// once — and scattered into LDS in A-fragment layout. 8 waves, each wave
// owns ntiles {wave, wave+8, wave+16, wave+24}.
// ---------------------------------------------------------------------------
__global__ __launch_bounds__(256, 2) void istft_gemm(const float* __restrict__ xf,
                                                     const _Float16* __restrict__ basis,
                                                     float* __restrict__ out) {
    __shared__ __align__(32) _Float16 lds_a[KSTEPS * 512];   // 8704 halves

    const int tid    = threadIdx.x;
    const int m_base = blockIdx.x * 16;
    const int bb     = m_base / TT;          // block-uniform batch index
    const int t_base = m_base - bb * TT;     // block-uniform frame index

    // --- zero the K-padding kstep (K = 512..543 live in kstep 16) ---
    {
        int z = 16 * 512 + tid * 2;
        lds_a[z]     = (_Float16)0.f;
        lds_a[z + 1] = (_Float16)0.f;
    }
    __syncthreads();

    // --- cooperative A-tile fill: 257 freqs x 16 rows as float2 loads ---
    // idx -> (f = idx>>4, M = idx&15); lanes step M => 16 lanes x 8B = 128B.
    for (int idx = tid; idx < FF * 16; idx += 256) {
        int M = idx & 15;
        int f = idx >> 4;
        const float2 v = ((const float2*)xf)[(size_t)(bb * FF + f) * TT + (t_base + M)];
        // scatter: real -> K=f, imag -> K=f+257, via inverse fragment map
        // K = kstep*32 + r ; h = (r&7) + ((r&16)?8:0) ; lane = M + 16*((r>>3)&1)
        {
            int K = f;
            int ks = K >> 5, r = K & 31;
            int h = (r & 7) + ((r & 16) ? 8 : 0);
            int lane = M + (((r >> 3) & 1) << 4);
            lds_a[ks * 512 + lane * 16 + h] = (_Float16)v.x;
        }
        {
            int K = f + FF;
            int ks = K >> 5, r = K & 31;
            int h = (r & 7) + ((r & 16) ? 8 : 0);
            int lane = M + (((r >> 3) & 1) << 4);
            lds_a[ks * 512 + lane * 16 + h] = (_Float16)v.y;
        }
    }
    __syncthreads();

    const int wave = tid >> 5;
    const int lane = tid & 31;

    v8f acc0 = {}, acc1 = {}, acc2 = {}, acc3 = {};
    const size_t nt_stride = (size_t)KSTEPS * 512;           // halves per ntile

    for (int ks = 0; ks < KSTEPS; ++ks) {
        v16h a = *(const v16h*)(&lds_a[ks * 512 + lane * 16]);
        const _Float16* bp = basis + ((size_t)ks * 32 + lane) * 16;
        v16h b0 = *(const v16h*)(bp + (size_t)(wave)      * nt_stride);
        v16h b1 = *(const v16h*)(bp + (size_t)(wave +  8) * nt_stride);
        v16h b2 = *(const v16h*)(bp + (size_t)(wave + 16) * nt_stride);
        v16h b3 = *(const v16h*)(bp + (size_t)(wave + 24) * nt_stride);
        acc0 = __builtin_amdgcn_wmma_f32_16x16x32_f16(false, a, false, b0, (short)0, acc0, false, false);
        acc1 = __builtin_amdgcn_wmma_f32_16x16x32_f16(false, a, false, b1, (short)0, acc1, false, false);
        acc2 = __builtin_amdgcn_wmma_f32_16x16x32_f16(false, a, false, b2, (short)0, acc2, false, false);
        acc3 = __builtin_amdgcn_wmma_f32_16x16x32_f16(false, a, false, b3, (short)0, acc3, false, false);
    }

    // --- epilogue: overlap-add scatter, C/D frag: M = r + 8*(lane>=16), N = lane&15
    const int Nn   = lane & 15;
    const int Moff = (lane & 16) ? 8 : 0;
    float* brow = out + (size_t)bb * OUT_LEN + Nn;
#pragma unroll
    for (int r = 0; r < 8; ++r) {
        int t = t_base + Moff + r;
        float* row = brow + t * HOP;
        unsafeAtomicAdd(row + (wave)      * 16, acc0[r]);
        unsafeAtomicAdd(row + (wave +  8) * 16, acc1[r]);
        unsafeAtomicAdd(row + (wave + 16) * 16, acc2[r]);
        unsafeAtomicAdd(row + (wave + 24) * 16, acc3[r]);
    }
}

// ---------------------------------------------------------------------------
// Kernel 3: window-power normalization. Each sample s is covered by frames
// t = s/256 (n = s%256) and t = s/256 - 1 (n = s%256 + 256) when valid.
// ---------------------------------------------------------------------------
__global__ __launch_bounds__(256) void norm_kernel(const float* __restrict__ window,
                                                   float* __restrict__ out) {
    int idx = blockIdx.x * 256 + threadIdx.x;     // exact grid, no guard
    int b   = idx / OUT_LEN;
    int s   = idx - b * OUT_LEN;
    int q   = s >> 8;
    int r   = s & 255;
    float d = 0.f;
    if (q < TT) { float w = window[r];       d += w * w; }
    if (q >= 1) { float w = window[r + 256]; d += w * w; }   // q-1 < TT always
    out[idx] = out[idx] / (d + EPS_F);
}

// ---------------------------------------------------------------------------
extern "C" void kernel_launch(void* const* d_in, const int* in_sizes, int n_in,
                              void* d_out, int out_size, void* d_ws, size_t ws_size,
                              hipStream_t stream) {
    const float* xf     = (const float*)d_in[0];   // [16,257,4000,2] f32
    const float* window = (const float*)d_in[1];   // [512] f32
    float*    out   = (float*)d_out;               // [16, 1024256] f32
    _Float16* basis = (_Float16*)d_ws;             // 557 KB scratch

    // 16*OUT_LEN = 16388096 floats = 4097024 float4 = 16004 * 256
    zero_kernel<<<16004, 256, 0, stream>>>((v4f*)out);

    // 32 ntiles * 17 ksteps * 512 = 278528 = 1088 * 256
    basis_kernel<<<1088, 256, 0, stream>>>(window, basis);

    // 64000 rows / 16 = 4000 M-tiles
    istft_gemm<<<4000, 256, 0, stream>>>(xf, basis, out);

    // 16388096 / 256 = 64016
    norm_kernel<<<64016, 256, 0, stream>>>(window, out);
}
